// Scheduler_19868518711624
// MI455X (gfx1250) — matrix operand order repaired and checked by reading, not compiled, not run on hardware
//
#include <hip/hip_runtime.h>

typedef _Float16 f16;
typedef __attribute__((ext_vector_type(16))) _Float16 v16h;
typedef __attribute__((ext_vector_type(8)))  float    v8f;

#define CDIV(a,b) (((a)+(b)-1)/(b))

// ---------------------------------------------------------------- helpers
__device__ __forceinline__ float eluf(float x) { return x > 0.f ? x : expm1f(x); }
__device__ __forceinline__ float geluf(float x) {
  float t = 0.7978845608028654f * (x + 0.044715f * x * x * x);
  return 0.5f * x * (1.f + tanhf(t));
}
// order-preserving float<->uint map for atomicMax on floats
__device__ __forceinline__ unsigned ordf(float f) {
  unsigned u = __float_as_uint(f);
  return (u & 0x80000000u) ? ~u : (u | 0x80000000u);
}
__device__ __forceinline__ float unordf(unsigned u) {
  unsigned v = (u & 0x80000000u) ? (u & 0x7fffffffu) : ~u;
  return __uint_as_float(v);
}

// WMMA A fragment (16x32 f16) from row-major LDS tile; lowers to 2x ds_load_b128
__device__ __forceinline__ v16h load_a_frag(const f16* base, int ld) {
  int lane = threadIdx.x & 31;
  int m = lane & 15, g = lane >> 4;
  v16h a;
#pragma unroll
  for (int e = 0; e < 16; ++e) {
    int i = e >> 1, j = e & 1;
    int k = (i < 4) ? (2 * i + 8 * g + j) : (16 + 2 * (i - 4) + 8 * g + j);
    a[e] = base[(size_t)m * ld + k];
  }
  return a;
}

// ---------------------------------------------------------------- tiny utils
__global__ void fill_f32_k(float* p, float v, int n) {
  int i = blockIdx.x * blockDim.x + threadIdx.x;
  if (i < n) p[i] = v;
}
__global__ void fill_u32_k(unsigned* p, unsigned v, int n) {
  int i = blockIdx.x * blockDim.x + threadIdx.x;
  if (i < n) p[i] = v;
}
__global__ void conv_f16_k(const float* x, f16* y, int n) {
  int i = blockIdx.x * blockDim.x + threadIdx.x;
  if (i < n) y[i] = (f16)x[i];
}
__global__ void gelu_conv_k(const float* x, f16* y, int n) {
  int i = blockIdx.x * blockDim.x + threadIdx.x;
  if (i < n) y[i] = (f16)geluf(x[i]);
}

// Convert f32 weight [K,N] into fragment-ready f16 layout:
// chunk (kb,nb) = 32x16 tile; within chunk, element (lane,e) is contiguous so a
// wave's whole B fragment is one 32B-per-lane contiguous load.
__global__ void swizzleB_k(const float* __restrict__ W, f16* __restrict__ out, int K, int N) {
  int idx = blockIdx.x * blockDim.x + threadIdx.x;
  if (idx >= K * N) return;
  int chunk = idx >> 9;
  int within = idx & 511;
  int lane = within >> 4, e = within & 15;
  int nb16 = N >> 4;
  int kb = chunk / nb16, nb = chunk % nb16;
  int k = kb * 32 + 8 * (lane >> 4) + ((e < 8) ? e : (16 + (e - 8)));
  int n = nb * 16 + (lane & 15);
  out[idx] = (f16)W[(size_t)k * N + n];
}

// ---------------------------------------------------------------- generic WMMA GEMM
// C[M,N] = A[M,K](f16 row-major) * Bsw[K,N](f16, fragment-swizzled) (+bias).
// M%64==0, N%32==0, K%32==0. A double-buffer staged via CDNA5 async DMA so the
// global->LDS copy of tile i+1 overlaps the WMMA on tile i.
__global__ __launch_bounds__(256)
void gemm_f16_k(const f16* __restrict__ A, int lda,
                const f16* __restrict__ Bsw,
                const float* __restrict__ bias,
                float* __restrict__ C, int ldc,
                int M, int N, int K) {
  __shared__ f16 As[2][64 * 32];
  int m0 = blockIdx.y * 64, n0 = blockIdx.x * 32;
  int tid = threadIdx.x;
  int wave = tid >> 5, lane = tid & 31;
  int wm = wave & 3, wn = wave >> 2;  // 4 x 2 waves of 16x16 tiles
  int nb16 = N >> 4;
  int srow = tid >> 2, scc = tid & 3;  // A staging coords (16B per thread)
  unsigned ldsoff[2] = {(unsigned)(uintptr_t)(&As[0][srow * 32 + scc * 8]),
                        (unsigned)(uintptr_t)(&As[1][srow * 32 + scc * 8])};
  const f16* arow = A + (size_t)(m0 + srow) * lda + scc * 8;
  int KT = K >> 5;
  {
    unsigned long long ga = (unsigned long long)(uintptr_t)arow;
    asm volatile("global_load_async_to_lds_b128 %0, %1, off"
                 :: "v"(ldsoff[0]), "v"(ga) : "memory");
  }
  v8f acc = {};
  for (int ki = 0; ki < KT; ++ki) {
    if (ki + 1 < KT) {
      unsigned long long ga = (unsigned long long)(uintptr_t)(arow + (ki + 1) * 32);
      asm volatile("global_load_async_to_lds_b128 %0, %1, off"
                   :: "v"(ldsoff[(ki + 1) & 1]), "v"(ga) : "memory");
      asm volatile("s_wait_asynccnt 0x1" ::: "memory");
    } else {
      asm volatile("s_wait_asynccnt 0x0" ::: "memory");
    }
    __syncthreads();
    const f16* Ab = As[ki & 1];
    v16h a = load_a_frag(Ab + (wm * 16) * 32, 32);
    v16h b = *(const v16h*)(Bsw + ((size_t)ki * nb16 + (n0 >> 4) + wn) * 512 + lane * 16);
    acc = __builtin_amdgcn_wmma_f32_16x16x32_f16(false, a, false, b,
                                                 (short)0, acc, false, false);
    __syncthreads();
  }
  int n = n0 + wn * 16 + (lane & 15);
  int g = lane >> 4;
  float bv = bias ? bias[n] : 0.f;
#pragma unroll
  for (int r = 0; r < 8; ++r) {
    int m = m0 + wm * 16 + r + 8 * g;
    C[(size_t)m * ldc + n] = acc[r] + bv;
  }
}

// ---------------------------------------------------------------- HGT pieces
__global__ void relproj_k(const float* __restrict__ Ksrc, const float* __restrict__ Vsrc,
                          const float* __restrict__ arel, const float* __restrict__ mrel,
                          float* __restrict__ krel, float* __restrict__ vrel, int N) {
  int idx = blockIdx.x * blockDim.x + threadIdx.x;
  if (idx >= N * 256) return;
  int n = idx >> 8, h = (idx >> 5) & 7, dp = idx & 31;
  const float* kr = Ksrc + (size_t)n * 256 + h * 32;
  const float* vr = Vsrc + (size_t)n * 256 + h * 32;
  const float* am = arel + h * 1024 + dp;
  const float* mm = mrel + h * 1024 + dp;
  float sk = 0.f, sv = 0.f;
#pragma unroll 8
  for (int d = 0; d < 32; ++d) {
    sk += kr[d] * am[d * 32];
    sv += vr[d] * mm[d * 32];
  }
  krel[idx] = sk;
  vrel[idx] = sv;
}

__global__ void edge_alpha_k(const int* __restrict__ ei, int Er,
                             const float* __restrict__ Qd, const float* __restrict__ krel,
                             const float* __restrict__ prel, float invsq,
                             float* __restrict__ alpha, unsigned* __restrict__ amax) {
  int idx = blockIdx.x * blockDim.x + threadIdx.x;
  if (idx >= Er * 8) return;
  int e = idx >> 3, h = idx & 7;
  int s = ei[e], d = ei[Er + e];
  const float* q = Qd + (size_t)d * 256 + h * 32;
  const float* k = krel + (size_t)s * 256 + h * 32;
  float a = 0.f;
#pragma unroll 8
  for (int i = 0; i < 32; ++i) a += q[i] * k[i];
  a *= prel[h] * invsq;
  alpha[idx] = a;
  atomicMax(&amax[d * 8 + h], ordf(a));
}

__global__ void edge_exp_k(const int* __restrict__ ei, int Er,
                           const float* __restrict__ vrel, const float* __restrict__ alpha,
                           const unsigned* __restrict__ amax,
                           float* __restrict__ den, float* __restrict__ aggr) {
  int idx = blockIdx.x * blockDim.x + threadIdx.x;
  if (idx >= Er * 8) return;
  int e = idx >> 3, h = idx & 7;
  int s = ei[e], d = ei[Er + e];
  float ex = expf(alpha[idx] - unordf(amax[d * 8 + h]));
  atomicAdd(&den[d * 8 + h], ex);
  const float* v = vrel + (size_t)s * 256 + h * 32;
  float* o = aggr + (size_t)d * 256 + h * 32;
#pragma unroll 8
  for (int i = 0; i < 32; ++i) atomicAdd(&o[i], v[i] * ex);
}

__global__ void agg_merge_k(const float* __restrict__ aggr, const float* __restrict__ den,
                            float* __restrict__ agg, int N) {
  int idx = blockIdx.x * blockDim.x + threadIdx.x;
  if (idx >= N * 256) return;
  int n = idx >> 8, h = (idx >> 5) & 7;
  float dv = den[n * 8 + h];
  if (dv > 0.f) agg[idx] += aggr[idx] / dv;
}

__global__ void skip_elu_k(const float* __restrict__ o, const float* __restrict__ xprev,
                           const float* __restrict__ skipv, int use_skip,
                           float* __restrict__ xs, f16* __restrict__ xs16, int n) {
  int idx = blockIdx.x * blockDim.x + threadIdx.x;
  if (idx >= n) return;
  float v = o[idx];
  if (use_skip) {
    float g = 1.f / (1.f + expf(-skipv[0]));
    v = g * v + (1.f - g) * xprev[idx];
  }
  v = eluf(v);
  xs[idx] = v;
  xs16[idx] = (f16)v;
}

__global__ void pool_mean_k(const float* __restrict__ x, int n,
                            float* __restrict__ out, int outld, int outoff) {
  int idx = blockIdx.x * blockDim.x + threadIdx.x;
  if (idx >= 64 * 256) return;
  int b = idx >> 8, e = idx & 255;
  float s = 0.f;
  for (int i = 0; i < n; ++i) s += x[((size_t)b * n + i) * 256 + e];
  out[(size_t)b * outld + outoff + e] = s / (float)n;
}

// generic small dense (fp32, W row-major [K,N])
__global__ void dense_k(const float* __restrict__ in, int ldi,
                        const float* __restrict__ W, int ldw,
                        const float* __restrict__ bias,
                        float* __restrict__ out, int ldo,
                        int R, int N, int K, int act) {
  int idx = blockIdx.x * blockDim.x + threadIdx.x;
  if (idx >= R * N) return;
  int r = idx / N, j = idx % N;
  float s = bias ? bias[j] : 0.f;
  for (int k = 0; k < K; ++k) s += in[(size_t)r * ldi + k] * W[(size_t)k * ldw + j];
  if (act) s = eluf(s);
  out[(size_t)r * ldo + j] = s;
}

// ---------------------------------------------------------------- fused actor
// Per block: 128 action rows. h1 tile = 128x1024 f16 = 256KB of the 320KB/WGP
// CDNA5 LDS. 8 waves = 2(M) x 4(N); each wave computes 4 M-subtiles per B
// fragment (register reuse -> 4x less aw1 L2 traffic), then contracts h2 with
// aw2 into per-row logits held in registers. h1/h2 never touch HBM.
__global__ __launch_bounds__(256)
void actor_fused_k(const float* __restrict__ Ac, const float* __restrict__ Ap,
                   const float* __restrict__ Apool,
                   const f16* __restrict__ aw1sw, const float* __restrict__ ab1,
                   const float* __restrict__ aw2, const float* __restrict__ ab2,
                   float* __restrict__ logits) {
  __shared__ f16 h1s[128 * 1024];
  __shared__ float slog[128];
  int tid = threadIdx.x;
  int blk = blockIdx.x;
  int b = blk >> 3;  // 8 blocks per batch element (1024 rows / 128)
  if (tid < 128) slog[tid] = 0.f;
  // build h1 tile: float4 reads, packed 16B f16 stores
  for (int it = tid * 8; it < 128 * 1024; it += 256 * 8) {
    int r = it >> 10, j = it & 1023;
    int grow = blk * 128 + r;
    int c = grow & 3;
    int p = (grow & 1023) >> 2;
    const float* acp = Ac + (size_t)(b * 4 + c) * 1024 + j;
    const float* app = Ap + (size_t)(b * 256 + p) * 1024 + j;
    const float* plp = Apool + (size_t)b * 1024 + j;
    float4 a0 = *(const float4*)acp, a1 = *(const float4*)(acp + 4);
    float4 p0 = *(const float4*)app, p1 = *(const float4*)(app + 4);
    float4 q0 = *(const float4*)plp, q1 = *(const float4*)(plp + 4);
    f16 tmp[8];
    tmp[0] = (f16)eluf(a0.x + p0.x + q0.x);
    tmp[1] = (f16)eluf(a0.y + p0.y + q0.y);
    tmp[2] = (f16)eluf(a0.z + p0.z + q0.z);
    tmp[3] = (f16)eluf(a0.w + p0.w + q0.w);
    tmp[4] = (f16)eluf(a1.x + p1.x + q1.x);
    tmp[5] = (f16)eluf(a1.y + p1.y + q1.y);
    tmp[6] = (f16)eluf(a1.z + p1.z + q1.z);
    tmp[7] = (f16)eluf(a1.w + p1.w + q1.w);
    *(uint4*)(h1s + it) = *(const uint4*)tmp;
  }
  __syncthreads();
  int wave = tid >> 5, lane = tid & 31;
  int wm = wave & 1, wn = wave >> 1;  // rows: 2x64, cols: 4x16 per pass
  float racc[4][8];
#pragma unroll
  for (int s = 0; s < 4; ++s)
#pragma unroll
    for (int r = 0; r < 8; ++r) racc[s][r] = 0.f;
  for (int nc = 0; nc < 16; ++nc) {
    int n0 = nc * 64 + wn * 16;
    v8f acc0 = {}, acc1 = {}, acc2 = {}, acc3 = {};
    for (int kk = 0; kk < 32; ++kk) {
      v16h bf = *(const v16h*)(aw1sw + ((size_t)kk * 64 + (n0 >> 4)) * 512 + lane * 16);
      const f16* ab = h1s + (size_t)(wm * 64) * 1024 + kk * 32;
      v16h a0 = load_a_frag(ab, 1024);
      acc0 = __builtin_amdgcn_wmma_f32_16x16x32_f16(false, a0, false, bf, (short)0, acc0, false, false);
      v16h a1 = load_a_frag(ab + (size_t)16 * 1024, 1024);
      acc1 = __builtin_amdgcn_wmma_f32_16x16x32_f16(false, a1, false, bf, (short)0, acc1, false, false);
      v16h a2 = load_a_frag(ab + (size_t)32 * 1024, 1024);
      acc2 = __builtin_amdgcn_wmma_f32_16x16x32_f16(false, a2, false, bf, (short)0, acc2, false, false);
      v16h a3 = load_a_frag(ab + (size_t)48 * 1024, 1024);
      acc3 = __builtin_amdgcn_wmma_f32_16x16x32_f16(false, a3, false, bf, (short)0, acc3, false, false);
    }
    int n = n0 + (lane & 15);
    float b1 = ab1[n], w2 = aw2[n];
#pragma unroll
    for (int r = 0; r < 8; ++r) {
      racc[0][r] += eluf(acc0[r] + b1) * w2;
      racc[1][r] += eluf(acc1[r] + b1) * w2;
      racc[2][r] += eluf(acc2[r] + b1) * w2;
      racc[3][r] += eluf(acc3[r] + b1) * w2;
    }
  }
  int g = lane >> 4;
#pragma unroll
  for (int s = 0; s < 4; ++s)
#pragma unroll
    for (int r = 0; r < 8; ++r)
      atomicAdd(&slog[wm * 64 + s * 16 + r + 8 * g], racc[s][r]);
  __syncthreads();
  if (tid < 128) logits[(size_t)blk * 128 + tid] = slog[tid] + ab2[0];
}

// ---------------------------------------------------------------- softmax/gather
__global__ void softmax_gather_k(const float* __restrict__ logits,
                                 const unsigned char* __restrict__ mask,
                                 const int* __restrict__ action,
                                 float* __restrict__ out) {
  __shared__ float red[256];
  int b = blockIdx.x, tid = threadIdx.x;
  float mx = -1e30f;
  for (int i = tid; i < 1024; i += 256) {
    int p = i >> 2, c = i & 3;
    float lg = mask[b * 1024 + c * 256 + p] ? logits[(size_t)b * 1024 + i] : -1e9f;
    mx = fmaxf(mx, lg);
  }
  red[tid] = mx;
  __syncthreads();
  for (int s = 128; s > 0; s >>= 1) {
    if (tid < s) red[tid] = fmaxf(red[tid], red[tid + s]);
    __syncthreads();
  }
  mx = red[0];
  __syncthreads();
  float sum = 0.f;
  for (int i = tid; i < 1024; i += 256) {
    int p = i >> 2, c = i & 3;
    float lg = mask[b * 1024 + c * 256 + p] ? logits[(size_t)b * 1024 + i] : -1e9f;
    sum += expf(lg - mx);
  }
  red[tid] = sum;
  __syncthreads();
  for (int s = 128; s > 0; s >>= 1) {
    if (tid < s) red[tid] += red[tid + s];
    __syncthreads();
  }
  if (tid == 0) {
    float lse = mx + logf(red[0]);
    int a = action[b];
    int p = a >> 2, c = a & 3;
    float lg = mask[b * 1024 + c * 256 + p] ? logits[(size_t)b * 1024 + a] : -1e9f;
    out[b] = lg - lse;
  }
}

// ---------------------------------------------------------------- host driver
extern "C" void kernel_launch(void* const* d_in, const int* in_sizes, int n_in,
                              void* d_out, int out_size, void* d_ws, size_t ws_size,
                              hipStream_t stream) {
  const float* x_crane = (const float*)d_in[0];
  const float* x_pile  = (const float*)d_in[1];
  const int* eis[4] = {(const int*)d_in[2], (const int*)d_in[3],
                       (const int*)d_in[4], (const int*)d_in[5]};
  int Ers[4] = {in_sizes[2] / 2, in_sizes[3] / 2, in_sizes[4] / 2, in_sizes[5] / 2};
  int stt[4] = {0, 1, 0, 1}, dtt[4] = {0, 0, 1, 1};
  const int* batch_action = (const int*)d_in[6];
  const unsigned char* batch_mask = (const unsigned char*)d_in[7];
  const float* kqv0_w = (const float*)d_in[8];
  const float* kqv0_b = (const float*)d_in[9];
  const float* kqv1_w = (const float*)d_in[10];
  const float* kqv1_b = (const float*)d_in[11];
  const float* a_rel = (const float*)d_in[12];
  const float* m_rel = (const float*)d_in[13];
  const float* p_rel = (const float*)d_in[14];
  const float* out_w = (const float*)d_in[15];
  const float* out_b = (const float*)d_in[16];
  const float* skipw = (const float*)d_in[17];
  const float* aw0 = (const float*)d_in[18];
  const float* ab0 = (const float*)d_in[19];
  const float* aw1 = (const float*)d_in[20];
  const float* ab1 = (const float*)d_in[21];
  const float* aw2 = (const float*)d_in[22];
  const float* ab2 = (const float*)d_in[23];
  const float* cw0 = (const float*)d_in[24];
  const float* cb0 = (const float*)d_in[25];
  const float* cw1 = (const float*)d_in[26];
  const float* cb1 = (const float*)d_in[27];
  const float* cw2 = (const float*)d_in[28];
  const float* cb2 = (const float*)d_in[29];
  float* out = (float*)d_out;

  const int NtC = 256, NtP = 16384;
  size_t off = 0;
  char* base = (char*)d_ws;
  auto alloc = [&](size_t bytes) -> void* {
    void* p = base + off;
    off += (bytes + 255) & ~size_t(255);
    return p;
  };
  // fragment-swizzled f16 weights
  f16* kqv0w16 = (f16*)alloc((size_t)2 * 3 * 64 * 256 * 2);
  f16* kqv1w16 = (f16*)alloc((size_t)2 * 3 * 256 * 256 * 2);
  f16* outw16  = (f16*)alloc((size_t)2 * 2 * 256 * 256 * 2);
  f16* aw0swA  = (f16*)alloc((size_t)256 * 1024 * 2);   // aw0 rows 0..255
  f16* aw0swB  = (f16*)alloc((size_t)256 * 1024 * 2);   // aw0 rows 256..511
  f16* aw1sw   = (f16*)alloc((size_t)1024 * 1024 * 2);
  f16* xc16in  = (f16*)alloc((size_t)NtC * 64 * 2);
  f16* xp16in  = (f16*)alloc((size_t)NtP * 64 * 2);
  float* KQV[3][2];
  for (int j = 0; j < 3; ++j) {
    KQV[j][0] = (float*)alloc((size_t)NtC * 256 * 4);
    KQV[j][1] = (float*)alloc((size_t)NtP * 256 * 4);
  }
  float* krel = (float*)alloc((size_t)NtP * 256 * 4);
  float* vrel = (float*)alloc((size_t)NtP * 256 * 4);
  float* agg[2] = {(float*)alloc((size_t)NtC * 256 * 4),
                   (float*)alloc((size_t)NtP * 256 * 4)};
  unsigned* amax = (unsigned*)alloc((size_t)NtP * 8 * 4);
  float* den  = (float*)alloc((size_t)NtP * 8 * 4);
  float* aggr = (float*)alloc((size_t)NtP * 256 * 4);
  float* alpha = (float*)alloc((size_t)262144 * 8 * 4);
  float* xs0[2] = {(float*)alloc((size_t)NtC * 256 * 4), (float*)alloc((size_t)NtP * 256 * 4)};
  float* xs1[2] = {(float*)alloc((size_t)NtC * 256 * 4), (float*)alloc((size_t)NtP * 256 * 4)};
  f16* xs0h[2] = {(f16*)alloc((size_t)NtC * 256 * 2), (f16*)alloc((size_t)NtP * 256 * 2)};
  f16* xs1h[2] = {(f16*)alloc((size_t)NtC * 256 * 2), (f16*)alloc((size_t)NtP * 256 * 2)};
  f16* g16[2] = {(f16*)alloc((size_t)NtC * 256 * 2), (f16*)alloc((size_t)NtP * 256 * 2)};
  float* obuf[2] = {(float*)alloc((size_t)NtC * 256 * 4), (float*)alloc((size_t)NtP * 256 * 4)};
  float* hpool = (float*)alloc(64 * 512 * 4);
  float* Apool = (float*)alloc(64 * 1024 * 4);
  float* Ac = (float*)alloc((size_t)256 * 1024 * 4);
  float* Ap = (float*)alloc((size_t)16384 * 1024 * 4);
  float* logits = (float*)alloc((size_t)65536 * 4);
  float* c1 = (float*)alloc(64 * 512 * 4);
  float* c2 = (float*)alloc(64 * 512 * 4);

  const int T = 256;
  auto g1 = [&](int n) { return dim3(CDIV(n, T)); };

  // weight swizzles (per weight-matrix slice) + input conversions
  for (int t = 0; t < 2; ++t)
    for (int j = 0; j < 3; ++j) {
      swizzleB_k<<<g1(64 * 256), T, 0, stream>>>(
          kqv0_w + (size_t)(t * 3 + j) * 64 * 256,
          kqv0w16 + (size_t)(t * 3 + j) * 64 * 256, 64, 256);
      swizzleB_k<<<g1(256 * 256), T, 0, stream>>>(
          kqv1_w + (size_t)(t * 3 + j) * 256 * 256,
          kqv1w16 + (size_t)(t * 3 + j) * 256 * 256, 256, 256);
    }
  for (int s = 0; s < 4; ++s)
    swizzleB_k<<<g1(256 * 256), T, 0, stream>>>(
        out_w + (size_t)s * 256 * 256, outw16 + (size_t)s * 256 * 256, 256, 256);
  swizzleB_k<<<g1(256 * 1024), T, 0, stream>>>(aw0, aw0swA, 256, 1024);
  swizzleB_k<<<g1(256 * 1024), T, 0, stream>>>(aw0 + (size_t)256 * 1024, aw0swB, 256, 1024);
  swizzleB_k<<<g1(1024 * 1024), T, 0, stream>>>(aw1, aw1sw, 1024, 1024);
  conv_f16_k<<<g1(NtC * 64), T, 0, stream>>>(x_crane, xc16in, NtC * 64);
  conv_f16_k<<<g1(NtP * 64), T, 0, stream>>>(x_pile, xp16in, NtP * 64);

  const int Nt[2] = {NtC, NtP};
  const float invsq = 0.17677669529663689f;  // 1/sqrt(32)

  for (int l = 0; l < 2; ++l) {
    int Kdim = (l == 0) ? 64 : 256;
    const f16* w16 = (l == 0) ? kqv0w16 : kqv1w16;
    const float* kqvb = (l == 0) ? kqv0_b : kqv1_b;
    const f16* xin16[2];
    const float* xprev[2];
    if (l == 0) { xin16[0] = xc16in; xin16[1] = xp16in; xprev[0] = x_crane; xprev[1] = x_pile; }
    else        { xin16[0] = xs0h[0]; xin16[1] = xs0h[1]; xprev[0] = xs0[0]; xprev[1] = xs0[1]; }
    float** xsout = (l == 0) ? xs0 : xs1;
    f16**  xsouth = (l == 0) ? xs0h : xs1h;

    // K/Q/V projections (WMMA GEMM, double-buffered async-DMA A, swizzled B)
    for (int t = 0; t < 2; ++t)
      for (int j = 0; j < 3; ++j)
        gemm_f16_k<<<dim3(256 / 32, Nt[t] / 64), T, 0, stream>>>(
            xin16[t], Kdim, w16 + (size_t)(t * 3 + j) * Kdim * 256,
            kqvb + (t * 3 + j) * 256, KQV[j][t], 256, Nt[t], 256, Kdim);

    fill_f32_k<<<g1(NtC * 256), T, 0, stream>>>(agg[0], 0.f, NtC * 256);
    fill_f32_k<<<g1(NtP * 256), T, 0, stream>>>(agg[1], 0.f, NtP * 256);

    for (int r = 0; r < 4; ++r) {
      int st = stt[r], dt = dtt[r], Ns = Nt[st], Nd = Nt[dt], Er = Ers[r];
      const float* arel = a_rel + (size_t)(l * 4 + r) * 8192;
      const float* mrel = m_rel + (size_t)(l * 4 + r) * 8192;
      const float* prel = p_rel + (l * 4 + r) * 8;
      relproj_k<<<g1(Ns * 256), T, 0, stream>>>(KQV[0][st], KQV[2][st], arel, mrel,
                                                krel, vrel, Ns);
      fill_u32_k<<<g1(Nd * 8), T, 0, stream>>>(amax, 0u, Nd * 8);
      fill_f32_k<<<g1(Nd * 8), T, 0, stream>>>(den, 0.f, Nd * 8);
      fill_f32_k<<<g1(Nd * 256), T, 0, stream>>>(aggr, 0.f, Nd * 256);
      edge_alpha_k<<<g1(Er * 8), T, 0, stream>>>(eis[r], Er, KQV[1][dt], krel, prel,
                                                 invsq, alpha, amax);
      edge_exp_k<<<g1(Er * 8), T, 0, stream>>>(eis[r], Er, vrel, alpha, amax, den, aggr);
      agg_merge_k<<<g1(Nd * 256), T, 0, stream>>>(aggr, den, agg[dt], Nd);
    }

    // output projection + skip + ELU
    for (int t = 0; t < 2; ++t) {
      gelu_conv_k<<<g1(Nt[t] * 256), T, 0, stream>>>(agg[t], g16[t], Nt[t] * 256);
      gemm_f16_k<<<dim3(256 / 32, Nt[t] / 64), T, 0, stream>>>(
          g16[t], 256, outw16 + (size_t)(l * 2 + t) * 256 * 256,
          out_b + (l * 2 + t) * 256, obuf[t], 256, Nt[t], 256, 256);
      skip_elu_k<<<g1(Nt[t] * 256), T, 0, stream>>>(obuf[t], xprev[t],
                                                    skipw + (l * 2 + t), l > 0,
                                                    xsout[t], xsouth[t], Nt[t] * 256);
    }
  }

  // pooled features: hpool = [hc_pool | hp_pool]  (64 x 512)
  pool_mean_k<<<g1(64 * 256), T, 0, stream>>>(xs1[0], 4, hpool, 512, 0);
  pool_mean_k<<<g1(64 * 256), T, 0, stream>>>(xs1[1], 256, hpool, 512, 256);

  // actor layer-1 decomposition
  gemm_f16_k<<<dim3(1024 / 32, 256 / 64), T, 0, stream>>>(
      xs1h[0], 256, aw0swA, nullptr, Ac, 1024, 256, 1024, 256);
  gemm_f16_k<<<dim3(1024 / 32, 16384 / 64), T, 0, stream>>>(
      xs1h[1], 256, aw0swB, nullptr, Ap, 1024, 16384, 1024, 256);
  dense_k<<<g1(64 * 1024), T, 0, stream>>>(hpool, 512, aw0 + (size_t)512 * 1024, 1024,
                                           ab0, Apool, 1024, 64, 1024, 512, 0);

  // fused actor: 128-row h1 tile in LDS -> WMMA h2 (4x B-fragment reuse) -> logits
  actor_fused_k<<<512, T, 0, stream>>>(Ac, Ap, Apool, aw1sw, ab1, aw2, ab2, logits);

  // masked log-softmax + action gather -> out[0..63]
  softmax_gather_k<<<64, T, 0, stream>>>(logits, batch_mask, batch_action, out);

  // critic MLP -> out[64..127]
  dense_k<<<g1(64 * 512), T, 0, stream>>>(hpool, 512, cw0, 512, cb0, c1, 512, 64, 512, 512, 1);
  dense_k<<<g1(64 * 512), T, 0, stream>>>(c1, 512, cw1, 512, cb1, c2, 512, 64, 512, 512, 1);
  dense_k<<<g1(64 * 1), T, 0, stream>>>(c2, 512, cw2, 1, cb2, out + 64, 1, 64, 1, 512, 0);
}
